// SharedMoELayer_15496242004513
// MI455X (gfx1250) — matrix operand order repaired and checked by reading, not compiled, not run on hardware
//
#include <hip/hip_runtime.h>
#include <hip/hip_bf16.h>
#include <stdint.h>

#define DIM 1024
#define INTER 4096
#define NUM_EXPERTS 8
#define NUM_TOKENS 8192
#define MTILE 32                        /* tokens per workgroup (2 WMMA M-subtiles) */
#define FCHUNK 128
#define TILES_PER (NUM_TOKENS / MTILE)  /* 256 */

typedef __attribute__((ext_vector_type(16))) __bf16 v16bf;
typedef __attribute__((ext_vector_type(8)))  float  v8f;

union FragU { uint4 q[2]; v16bf v; };

// ---------------------------------------------------------------------------
// Zero the per-expert token counters.
__global__ void init_kernel(int* cnt) {
    if (threadIdx.x < NUM_EXPERTS) cnt[threadIdx.x] = 0;
}

// ---------------------------------------------------------------------------
// Convert a [K,N] fp32 matrix into bf16 WMMA B-fragment layout:
// block b = ntile*(K/32)+kstep holds a 16(N)x32(K) tile; lane L stores 16
// contiguous-K bf16 values: n = ntile*16 + (L&15), k = kstep*32 + 16*(L>>4)+e.
__global__ __launch_bounds__(256) void swizzle_kernel(
        const float* __restrict__ src, __bf16* __restrict__ dst, int K, int N) {
    int t = blockIdx.x * blockDim.x + threadIdx.x;
    int kblocks = K >> 5;
    int total = kblocks * (N >> 4) * 32;
    if (t >= total) return;
    int lane = t & 31, b = t >> 5;
    int nt = b / kblocks, ks = b % kblocks;
    int n  = nt * 16 + (lane & 15);
    int kb = ks * 32 + 16 * (lane >> 4);
    __bf16* o = dst + (size_t)b * 512 + lane * 16;
#pragma unroll
    for (int e = 0; e < 16; ++e)
        o[e] = (__bf16)src[(size_t)(kb + e) * N + n];
}

// ---------------------------------------------------------------------------
// Router: one wave per token. logits = x @ Rw, top-2, softmax over the pair,
// scatter (token, gate) into per-expert lists with atomics.
__global__ __launch_bounds__(256) void router_kernel(
        const float* __restrict__ x, const float* __restrict__ rw,
        int* cnt, int* tok, float* gate) {
    int wv   = (blockIdx.x * blockDim.x + threadIdx.x) >> 5;
    int lane = threadIdx.x & 31;
    if (wv >= NUM_TOKENS) return;
    const float* xr = x + (size_t)wv * DIM;
    float acc[NUM_EXPERTS];
#pragma unroll
    for (int e = 0; e < NUM_EXPERTS; ++e) acc[e] = 0.f;
    for (int d = lane; d < DIM; d += 32) {
        float xv = xr[d];
        const float* r = rw + (size_t)d * NUM_EXPERTS;
#pragma unroll
        for (int e = 0; e < NUM_EXPERTS; ++e) acc[e] += xv * r[e];
    }
#pragma unroll
    for (int e = 0; e < NUM_EXPERTS; ++e)
        for (int off = 16; off > 0; off >>= 1)
            acc[e] += __shfl_xor(acc[e], off, 32);
    if (lane == 0) {
        int e0 = 0; float v0 = acc[0];
        for (int e = 1; e < NUM_EXPERTS; ++e) if (acc[e] > v0) { v0 = acc[e]; e0 = e; }
        int e1 = -1; float v1 = -3.4e38f;
        for (int e = 0; e < NUM_EXPERTS; ++e)
            if (e != e0 && acc[e] > v1) { v1 = acc[e]; e1 = e; }
        float w1 = 1.f / (1.f + __expf(v0 - v1));  // softmax over {v0, v1}
        float w0 = 1.f - w1;
        int p0 = atomicAdd(&cnt[e0], 1);
        tok[e0 * NUM_TOKENS + p0]  = wv;  gate[e0 * NUM_TOKENS + p0] = w0;
        int p1 = atomicAdd(&cnt[e1], 1);
        tok[e1 * NUM_TOKENS + p1]  = wv;  gate[e1 * NUM_TOKENS + p1] = w1;
    }
}

// ---------------------------------------------------------------------------
// Fused FFN:  out[tile] (+)= gate * ( relu(X[tile] @ W1) @ W2 )
// One workgroup = 32 tokens (two 16-row M-subtiles), 8 waves. F chunked by 128:
//   GEMM1: wave w computes H[0:32, fc+16w .. +16) over K=1024; each B fragment
//          feeds 2 WMMAs (both M-subtiles).
//   GEMM2: wave w owns out columns [128w,128w+128): 4 Ksteps x 8 Ntiles x 2 M.
// Weights are pre-swizzled bf16 fragments (pure b128 loads in the hot loop).
__global__ __launch_bounds__(256) void ffn_kernel(
        const float*  __restrict__ X,
        const __bf16* __restrict__ W1base,   // swizzled [K=DIM,  N=INTER] (xE)
        const __bf16* __restrict__ W2base,   // swizzled [K=INTER,N=DIM ] (xE)
        const int*    __restrict__ cnt,
        const int*    __restrict__ tokl,
        const float*  __restrict__ gatel,
        float* __restrict__ out, int routed) {
    int e, tile;
    if (routed) { e = blockIdx.x / TILES_PER; tile = blockIdx.x % TILES_PER; }
    else        { e = 0;                      tile = blockIdx.x; }
    int nTok = MTILE;
    if (routed) {
        nTok = cnt[e] - tile * MTILE;
        if (nTok <= 0) return;            // uniform per-block: EXEC stays full
        if (nTok > MTILE) nTok = MTILE;
    }
    const __bf16* W1 = W1base + (size_t)e * DIM * INTER;
    const __bf16* W2 = W2base + (size_t)e * DIM * INTER;

    __shared__ __bf16 Xs[MTILE][DIM + 8];      // +8 bf16 pad: kills bank conflicts
    __shared__ __bf16 Hs[MTILE][FCHUNK + 8];
    __shared__ float  gs[MTILE];
    __shared__ int    ts[MTILE];

    int tid = threadIdx.x;
    if (tid < MTILE) {
        int t; float g;
        if (routed) {
            if (tid < nTok) {
                t = tokl[e * NUM_TOKENS + tile * MTILE + tid];
                g = gatel[e * NUM_TOKENS + tile * MTILE + tid];
            } else { t = 0; g = 0.f; }     // padded row contributes 0
        } else { t = tile * MTILE + tid; g = 1.f; }
        ts[tid] = t; gs[tid] = g;
    }
    __syncthreads();
    for (int i = tid; i < MTILE * DIM; i += 256) {
        int r = i >> 10, c = i & (DIM - 1);
        Xs[r][c] = (__bf16)X[(size_t)ts[r] * DIM + c];
    }
    __syncthreads();

    int wave  = tid >> 5, lane = tid & 31;
    int half  = lane >> 4, ln = lane & 15;

    v8f out0[8], out1[8];
#pragma unroll
    for (int nt = 0; nt < 8; ++nt) {
        out0[nt] = (v8f){0,0,0,0,0,0,0,0};
        out1[nt] = (v8f){0,0,0,0,0,0,0,0};
    }

    const int KS1 = DIM / 32;     // 32 k-steps for GEMM1
    const int KSF = INTER / 32;   // 128 k-blocks in swizzled W2

    for (int fc = 0; fc < INTER; fc += FCHUNK) {
        // ---- GEMM1: H = relu(X @ W1[:, fc+16w .. +16)) --------------------
        v8f h0 = (v8f){0,0,0,0,0,0,0,0};
        v8f h1 = (v8f){0,0,0,0,0,0,0,0};
        const __bf16* w1p = W1 + ((size_t)((fc >> 4) + wave) * KS1) * 512 + lane * 16;
        for (int ks = 0; ks < KS1; ++ks) {
            FragU a0, a1, b;
            int kb = ks * 32 + 16 * half;                // A: one 16-K run/lane
            a0.q[0] = *(const uint4*)&Xs[ln][kb];
            a0.q[1] = *(const uint4*)&Xs[ln][kb + 8];
            a1.q[0] = *(const uint4*)&Xs[16 + ln][kb];
            a1.q[1] = *(const uint4*)&Xs[16 + ln][kb + 8];
            b.q[0]  = *(const uint4*)(w1p);
            b.q[1]  = *(const uint4*)(w1p + 8);
            w1p += 512;
            h0 = __builtin_amdgcn_wmma_f32_16x16x32_bf16(
                    false, a0.v, false, b.v, (short)0, h0, false, false);
            h1 = __builtin_amdgcn_wmma_f32_16x16x32_bf16(
                    false, a1.v, false, b.v, (short)0, h1, false, false);
        }
        __syncthreads();                                  // prev GEMM2 done with Hs
#pragma unroll
        for (int r = 0; r < 8; ++r) {
            float v0 = h0[r]; v0 = v0 > 0.f ? v0 : 0.f;   // ReLU
            float v1 = h1[r]; v1 = v1 > 0.f ? v1 : 0.f;
            Hs[r + 8 * half][wave * 16 + ln]      = (__bf16)v0;
            Hs[16 + r + 8 * half][wave * 16 + ln] = (__bf16)v1;
        }
        __syncthreads();

        // ---- GEMM2: out[:, 128w..] += H_chunk @ W2[fc..fc+128, :] ---------
        for (int ksub = 0; ksub < FCHUNK / 32; ++ksub) {
            FragU a0, a1;
            int kb = ksub * 32 + 16 * half;
            a0.q[0] = *(const uint4*)&Hs[ln][kb];
            a0.q[1] = *(const uint4*)&Hs[ln][kb + 8];
            a1.q[0] = *(const uint4*)&Hs[16 + ln][kb];
            a1.q[1] = *(const uint4*)&Hs[16 + ln][kb + 8];
            int ksg = (fc >> 5) + ksub;
#pragma unroll
            for (int nt = 0; nt < 8; ++nt) {
                const __bf16* w2p =
                    W2 + ((size_t)(wave * 8 + nt) * KSF + ksg) * 512 + lane * 16;
                FragU b;
                b.q[0] = *(const uint4*)(w2p);
                b.q[1] = *(const uint4*)(w2p + 8);
                out0[nt] = __builtin_amdgcn_wmma_f32_16x16x32_bf16(
                        false, a0.v, false, b.v, (short)0, out0[nt], false, false);
                out1[nt] = __builtin_amdgcn_wmma_f32_16x16x32_bf16(
                        false, a1.v, false, b.v, (short)0, out1[nt], false, false);
            }
        }
    }

    // ---- writeback: D-frag lane ln holds col, VGPR r holds row r+8*half ----
#pragma unroll
    for (int nt = 0; nt < 8; ++nt) {
        int col = wave * 128 + nt * 16 + ln;
#pragma unroll
        for (int r = 0; r < 8; ++r) {
            int m0 = r + 8 * half;
            int m1 = 16 + m0;
            float v0 = out0[nt][r] * gs[m0];
            float v1 = out1[nt][r] * gs[m1];
            float* dst0 = &out[(size_t)ts[m0] * DIM + col];
            float* dst1 = &out[(size_t)ts[m1] * DIM + col];
            if (routed) { atomicAdd(dst0, v0); atomicAdd(dst1, v1); }
            else        { *dst0 = v0;          *dst1 = v1; }
        }
    }
}

// ---------------------------------------------------------------------------
extern "C" void kernel_launch(void* const* d_in, const int* in_sizes, int n_in,
                              void* d_out, int out_size, void* d_ws, size_t ws_size,
                              hipStream_t stream) {
    (void)in_sizes; (void)n_in; (void)out_size; (void)ws_size;
    const float* X   = (const float*)d_in[0];
    const float* Ws1 = (const float*)d_in[1];
    const float* Ws2 = (const float*)d_in[2];
    const float* Wr1 = (const float*)d_in[3];
    const float* Wr2 = (const float*)d_in[4];
    const float* Rw  = (const float*)d_in[5];
    float* out = (float*)d_out;

    char* ws = (char*)d_ws;
    size_t off = 0;
    auto alloc = [&](size_t bytes) -> void* {
        void* p = (void*)(ws + off);
        off += (bytes + 255) & ~(size_t)255;
        return p;
    };
    int*    cnt  = (int*)   alloc(NUM_EXPERTS * sizeof(int));
    int*    tok  = (int*)   alloc((size_t)NUM_EXPERTS * NUM_TOKENS * sizeof(int));
    float*  gate = (float*) alloc((size_t)NUM_EXPERTS * NUM_TOKENS * sizeof(float));
    __bf16* W1s  = (__bf16*)alloc((size_t)DIM * INTER * 2);
    __bf16* W2s  = (__bf16*)alloc((size_t)DIM * INTER * 2);
    __bf16* W1r  = (__bf16*)alloc((size_t)NUM_EXPERTS * DIM * INTER * 2);
    __bf16* W2r  = (__bf16*)alloc((size_t)NUM_EXPERTS * DIM * INTER * 2);

    hipLaunchKernelGGL(init_kernel, dim3(1), dim3(64), 0, stream, cnt);

    int swblocks = ((DIM * INTER / 16) + 255) / 256;
    hipLaunchKernelGGL(swizzle_kernel, dim3(swblocks), dim3(256), 0, stream,
                       Ws1, W1s, DIM, INTER);
    hipLaunchKernelGGL(swizzle_kernel, dim3(swblocks), dim3(256), 0, stream,
                       Ws2, W2s, INTER, DIM);
    for (int e = 0; e < NUM_EXPERTS; ++e) {
        hipLaunchKernelGGL(swizzle_kernel, dim3(swblocks), dim3(256), 0, stream,
                           Wr1 + (size_t)e * DIM * INTER, W1r + (size_t)e * DIM * INTER,
                           DIM, INTER);
        hipLaunchKernelGGL(swizzle_kernel, dim3(swblocks), dim3(256), 0, stream,
                           Wr2 + (size_t)e * DIM * INTER, W2r + (size_t)e * DIM * INTER,
                           INTER, DIM);
    }

    hipLaunchKernelGGL(router_kernel, dim3(NUM_TOKENS / 8), dim3(256), 0, stream,
                       X, Rw, cnt, tok, gate);

    // Shared expert: writes every element of d_out (no init needed).
    hipLaunchKernelGGL(ffn_kernel, dim3(TILES_PER), dim3(256), 0, stream,
                       X, W1s, W2s, (const int*)nullptr, (const int*)nullptr,
                       (const float*)nullptr, out, 0);
    // Routed experts: gated atomic accumulation; empty tiles early-exit.
    hipLaunchKernelGGL(ffn_kernel, dim3(NUM_EXPERTS * TILES_PER), dim3(256), 0, stream,
                       X, W1r, W2r, cnt, tok, gate, out, 1);
}
